// BERTCPCLoss_2267742733132
// MI455X (gfx1250) — matrix operand order repaired and checked by reading, not compiled, not run on hardware
//
#include <hip/hip_runtime.h>

#define DEVFN static __device__ __forceinline__

typedef __attribute__((ext_vector_type(16))) __bf16 v16bf;
typedef __attribute__((ext_vector_type(8)))  float  v8f;

constexpr int Bn   = 32;
constexpr int Sn   = 512;
constexpr int Dn   = 1024;
constexpr int DROP = 128;
constexpr int KEEP = Sn - DROP;          // 384
constexpr int M    = Bn * DROP;          // 4096 rows of lossmat
constexpr int N    = Bn * Sn;            // 16384 cols of lossmat
constexpr int Kd   = Dn;                 // 1024 reduction dim
constexpr int BM   = 64;
constexpr int BN   = 128;
constexpr int NBLK = N / BN;             // 128 column blocks
constexpr int MSE_BLOCKS = 1024;

// ---- workspace layout (bytes) ----
constexpr size_t OFF_PHI  = 0;                                   // M*Kd*2  = 8 MB
constexpr size_t OFF_PLO  = OFF_PHI + (size_t)M * Kd * 2;        // 8 MB
constexpr size_t OFF_KHI  = OFF_PLO + (size_t)M * Kd * 2;        // 32 MB
constexpr size_t OFF_KLO  = OFF_KHI + (size_t)N * Kd * 2;        // 32 MB
constexpr size_t OFF_RMAX = OFF_KLO + (size_t)N * Kd * 2;        // M * f32 (final row max)
constexpr size_t OFF_RSUM = OFF_RMAX + (size_t)M * 4;            // M * f32 (final row sumexp)
constexpr size_t OFF_TLOG = OFF_RSUM + (size_t)M * 4;            // M * f32 (target logit)
constexpr size_t OFF_TCOL = OFF_TLOG + (size_t)M * 4;            // M * i32 (target column)
constexpr size_t OFF_MSEP = OFF_TCOL + (size_t)M * 4;            // MSE_BLOCKS * f32
constexpr size_t OFF_MPRT = OFF_MSEP + (size_t)MSE_BLOCKS * 4;   // M*NBLK f32 = 2 MB (partial max)
constexpr size_t OFF_SPRT = OFF_MPRT + (size_t)M * NBLK * 4;     // M*NBLK f32 = 2 MB (partial sumexp)

union Frag { uint4 u[2]; v16bf v; };

DEVFN unsigned short f2bf(float f) {             // round-to-nearest-even f32 -> bf16
  unsigned u = __float_as_uint(f);
  u += 0x7FFFu + ((u >> 16) & 1u);
  return (unsigned short)(u >> 16);
}
DEVFN float bf2f(unsigned short h) { return __uint_as_float(((unsigned)h) << 16); }

// ---------------- prep kernels ----------------
__global__ void convert_keys(const float* __restrict__ in_seq,
                             unsigned short* __restrict__ Khi,
                             unsigned short* __restrict__ Klo) {
  int idx = blockIdx.x * blockDim.x + threadIdx.x;  // float4 index
  constexpr int total4 = N * Kd / 4;                // 4,194,304
  if (idx >= total4) return;
  float4 v = ((const float4*)in_seq)[idx];
  unsigned short h0 = f2bf(v.x), h1 = f2bf(v.y), h2 = f2bf(v.z), h3 = f2bf(v.w);
  unsigned short l0 = f2bf(v.x - bf2f(h0)), l1 = f2bf(v.y - bf2f(h1));
  unsigned short l2 = f2bf(v.z - bf2f(h2)), l3 = f2bf(v.w - bf2f(h3));
  uint2 hv = make_uint2((unsigned)h0 | ((unsigned)h1 << 16), (unsigned)h2 | ((unsigned)h3 << 16));
  uint2 lv = make_uint2((unsigned)l0 | ((unsigned)l1 << 16), (unsigned)l2 | ((unsigned)l3 << 16));
  ((uint2*)Khi)[idx] = hv;
  ((uint2*)Klo)[idx] = lv;
}

__global__ void gather_preds(const float* __restrict__ out_seq,
                             const int* __restrict__ drop_idx,
                             unsigned short* __restrict__ Phi,
                             unsigned short* __restrict__ Plo,
                             int* __restrict__ tgt_col) {
  int m = blockIdx.x;                       // 0..M-1
  int b = m >> 7, j = m & (DROP - 1);
  int src = drop_idx[b * DROP + j];
  if (threadIdx.x == 0) tgt_col[m] = b * Sn + src;
  const float4* sp = (const float4*)(out_seq + ((size_t)b * Sn + src) * Dn);
  float4 v = sp[threadIdx.x];               // 256 threads * 4 = 1024 = Dn
  unsigned short h0 = f2bf(v.x), h1 = f2bf(v.y), h2 = f2bf(v.z), h3 = f2bf(v.w);
  unsigned short l0 = f2bf(v.x - bf2f(h0)), l1 = f2bf(v.y - bf2f(h1));
  unsigned short l2 = f2bf(v.z - bf2f(h2)), l3 = f2bf(v.w - bf2f(h3));
  uint2 hv = make_uint2((unsigned)h0 | ((unsigned)h1 << 16), (unsigned)h2 | ((unsigned)h3 << 16));
  uint2 lv = make_uint2((unsigned)l0 | ((unsigned)l1 << 16), (unsigned)l2 | ((unsigned)l3 << 16));
  ((uint2*)(Phi + (size_t)m * Dn))[threadIdx.x] = hv;
  ((uint2*)(Plo + (size_t)m * Dn))[threadIdx.x] = lv;
}

__global__ void mse_kernel(const float* __restrict__ in_seq,
                           const float* __restrict__ out_seq,
                           const int* __restrict__ keep_idx,
                           float* __restrict__ mse_part) {
  __shared__ float sm[256];
  constexpr int TOT4 = Bn * KEEP * Dn / 4;   // 3,145,728
  float acc = 0.f;
  for (int idx = blockIdx.x * blockDim.x + threadIdx.x; idx < TOT4;
       idx += gridDim.x * blockDim.x) {
    int e = idx * 4;
    int b = e / (KEEP * Dn);
    int rem = e - b * (KEEP * Dn);
    int r = rem >> 10;
    int d = rem & (Dn - 1);
    int kr = keep_idx[b * KEEP + r];
    size_t base = ((size_t)b * Sn + kr) * Dn + d;
    float4 vi = *(const float4*)(in_seq + base);
    float4 vo = *(const float4*)(out_seq + base);
    float dx = vi.x - vo.x, dy = vi.y - vo.y, dz = vi.z - vo.z, dw = vi.w - vo.w;
    acc += dx * dx + dy * dy + dz * dz + dw * dw;
  }
  sm[threadIdx.x] = acc; __syncthreads();
  for (int s = 128; s > 0; s >>= 1) {
    if (threadIdx.x < s) sm[threadIdx.x] += sm[threadIdx.x + s];
    __syncthreads();
  }
  if (threadIdx.x == 0) mse_part[blockIdx.x] = sm[0];
}

// ---------------- single-pass fused GEMM + per-block softmax partials ----------
// Each WG computes a 64x128 slab of lossmat (bf16x3 emulated fp32 WMMA), then
// emits per-(row, col-block) partial (max, sumexp) pairs + the target logit.
__global__ void __launch_bounds__(256)
gemm_softmax(const unsigned short* __restrict__ Phi,
             const unsigned short* __restrict__ Plo,
             const unsigned short* __restrict__ Khi,
             const unsigned short* __restrict__ Klo,
             float* __restrict__ mpart,
             float* __restrict__ spart,
             float* __restrict__ tgt_logit,
             const int* __restrict__ tgt_col) {
  __shared__ float smem[BM * BN];     // 32 KB
  __shared__ float red[256];
  __shared__ float rowm[BM];
  const int bx = blockIdx.x, by = blockIdx.y;
  const int tid = threadIdx.x;
  const int w = tid >> 5, lane = tid & 31;
  const int l15 = lane & 15, lhi = lane >> 4;     // lane half select
  const int wm = w & 1;                           // 2 M-positions
  const int wn = w >> 1;                          // 4 N-positions

  const int rowBase = by * BM + (wm << 5);
  const int colBase = bx * BN + (wn << 5);
  const int aRow0 = rowBase + l15,  aRow1 = aRow0 + 16;
  const int bRow0 = colBase + l15,  bRow1 = bRow0 + 16;

  const unsigned short* aHi0 = Phi + (size_t)aRow0 * Kd;
  const unsigned short* aLo0 = Plo + (size_t)aRow0 * Kd;
  const unsigned short* aHi1 = Phi + (size_t)aRow1 * Kd;
  const unsigned short* aLo1 = Plo + (size_t)aRow1 * Kd;
  const unsigned short* b0Hi = Khi + (size_t)bRow0 * Kd;
  const unsigned short* b0Lo = Klo + (size_t)bRow0 * Kd;
  const unsigned short* b1Hi = Khi + (size_t)bRow1 * Kd;
  const unsigned short* b1Lo = Klo + (size_t)bRow1 * Kd;

  v8f c00 = {}, c01 = {}, c10 = {}, c11 = {};
  for (int kk = 0; kk < Kd; kk += 32) {
    Frag ah0, al0, ah1, al1, bh0, bl0, bh1, bl1;
    // A frag: lane holds K chunks {kk+8*lhi..+7} and {kk+16+8*lhi..+7}
    const uint4* p;
    p = (const uint4*)(aHi0 + kk) + lhi;      ah0.u[0] = p[0]; ah0.u[1] = p[2];
    p = (const uint4*)(aLo0 + kk) + lhi;      al0.u[0] = p[0]; al0.u[1] = p[2];
    p = (const uint4*)(aHi1 + kk) + lhi;      ah1.u[0] = p[0]; ah1.u[1] = p[2];
    p = (const uint4*)(aLo1 + kk) + lhi;      al1.u[0] = p[0]; al1.u[1] = p[2];
    // B frag: lane holds K chunks {kk+8*lhi..+7} and {kk+16+8*lhi..+7}
    p = (const uint4*)(b0Hi + kk) + lhi;      bh0.u[0] = p[0]; bh0.u[1] = p[2];
    p = (const uint4*)(b0Lo + kk) + lhi;      bl0.u[0] = p[0]; bl0.u[1] = p[2];
    p = (const uint4*)(b1Hi + kk) + lhi;      bh1.u[0] = p[0]; bh1.u[1] = p[2];
    p = (const uint4*)(b1Lo + kk) + lhi;      bl1.u[0] = p[0]; bl1.u[1] = p[2];

    // bf16x3 emulated-fp32: hi*hi + hi*lo + lo*hi, 2x2 register tile
    c00 = __builtin_amdgcn_wmma_f32_16x16x32_bf16(false, ah0.v, false, bh0.v, (short)0, c00, false, false);
    c00 = __builtin_amdgcn_wmma_f32_16x16x32_bf16(false, ah0.v, false, bl0.v, (short)0, c00, false, false);
    c00 = __builtin_amdgcn_wmma_f32_16x16x32_bf16(false, al0.v, false, bh0.v, (short)0, c00, false, false);
    c01 = __builtin_amdgcn_wmma_f32_16x16x32_bf16(false, ah0.v, false, bh1.v, (short)0, c01, false, false);
    c01 = __builtin_amdgcn_wmma_f32_16x16x32_bf16(false, ah0.v, false, bl1.v, (short)0, c01, false, false);
    c01 = __builtin_amdgcn_wmma_f32_16x16x32_bf16(false, al0.v, false, bh1.v, (short)0, c01, false, false);
    c10 = __builtin_amdgcn_wmma_f32_16x16x32_bf16(false, ah1.v, false, bh0.v, (short)0, c10, false, false);
    c10 = __builtin_amdgcn_wmma_f32_16x16x32_bf16(false, ah1.v, false, bl0.v, (short)0, c10, false, false);
    c10 = __builtin_amdgcn_wmma_f32_16x16x32_bf16(false, al1.v, false, bh0.v, (short)0, c10, false, false);
    c11 = __builtin_amdgcn_wmma_f32_16x16x32_bf16(false, ah1.v, false, bh1.v, (short)0, c11, false, false);
    c11 = __builtin_amdgcn_wmma_f32_16x16x32_bf16(false, ah1.v, false, bl1.v, (short)0, c11, false, false);
    c11 = __builtin_amdgcn_wmma_f32_16x16x32_bf16(false, al1.v, false, bh1.v, (short)0, c11, false, false);
  }

  // C layout: VGPR r -> M = tileM + r + 8*lhi ; lane l15 -> N offset
  const int mloc0 = (wm << 5) + (lhi << 3);
  const int mloc1 = mloc0 + 16;
  const int nloc0 = (wn << 5) + l15;
  const int nloc1 = nloc0 + 16;
#pragma unroll
  for (int r = 0; r < 8; ++r) {
    smem[(mloc0 + r) * BN + nloc0] = c00[r];
    smem[(mloc0 + r) * BN + nloc1] = c01[r];
    smem[(mloc1 + r) * BN + nloc0] = c10[r];
    smem[(mloc1 + r) * BN + nloc1] = c11[r];
  }
  __syncthreads();

  // 4 threads per row, 32 cols each: row max, then sumexp vs row max
  const int r = tid >> 2;          // 0..63
  const int q = tid & 3;           // quarter
  const float* rowp = &smem[r * BN + q * 32];
  float mx = -3.4e38f;
  for (int c = 0; c < 32; ++c) mx = fmaxf(mx, rowp[c]);
  red[tid] = mx;
  __syncthreads();
  if (q == 0) {
    float m4 = fmaxf(fmaxf(red[tid], red[tid + 1]), fmaxf(red[tid + 2], red[tid + 3]));
    rowm[r] = m4;
  }
  __syncthreads();
  const float mrow = rowm[r];
  float sq = 0.f;
  for (int c = 0; c < 32; ++c) sq += __expf(rowp[c] - mrow);
  red[tid] = sq;
  __syncthreads();
  if (q == 0) {
    const int gm = by * BM + r;
    mpart[(size_t)gm * NBLK + bx] = mrow;
    spart[(size_t)gm * NBLK + bx] = red[tid] + red[tid + 1] + red[tid + 2] + red[tid + 3];
    int t = tgt_col[gm];
    int cbase = bx * BN;
    if (t >= cbase && t < cbase + BN) tgt_logit[gm] = smem[r * BN + (t - cbase)];
  }
}

// ---------------- deterministic log-sum-exp merge per row ----------------
__global__ void reduce_rows(const float* __restrict__ mpart,
                            const float* __restrict__ spart,
                            float* __restrict__ rowmax,
                            float* __restrict__ rowsum) {
  __shared__ float sm[NBLK];
  __shared__ float ss[NBLK];
  const int row = blockIdx.x;
  const int t = threadIdx.x;                 // NBLK threads
  const float mi = mpart[(size_t)row * NBLK + t];
  sm[t] = mi;
  __syncthreads();
  for (int s = NBLK / 2; s > 0; s >>= 1) {
    if (t < s) sm[t] = fmaxf(sm[t], sm[t + s]);
    __syncthreads();
  }
  const float gmax = sm[0];
  __syncthreads();
  ss[t] = spart[(size_t)row * NBLK + t] * __expf(mi - gmax);
  __syncthreads();
  for (int s = NBLK / 2; s > 0; s >>= 1) {
    if (t < s) ss[t] += ss[t + s];
    __syncthreads();
  }
  if (t == 0) { rowmax[row] = gmax; rowsum[row] = ss[0]; }
}

__global__ void finalize_kernel(const float* __restrict__ rowmax,
                                const float* __restrict__ rowsum,
                                const float* __restrict__ tgt_logit,
                                const float* __restrict__ mse_part,
                                float* __restrict__ out) {
  __shared__ float sx[256];
  __shared__ float sq[256];
  __shared__ int   sa[256];
  int tid = threadIdx.x;
  float xe = 0.f; int cnt = 0;
  for (int r = tid; r < M; r += 256) {
    float mrow = rowmax[r];
    float tl   = tgt_logit[r];
    xe  += mrow + __logf(rowsum[r]) - tl;   // -log_softmax at target
    cnt += (tl >= mrow) ? 1 : 0;            // target achieves the row max
  }
  float msum = 0.f;
  for (int i = tid; i < MSE_BLOCKS; i += 256) msum += mse_part[i];
  sx[tid] = xe; sq[tid] = msum; sa[tid] = cnt;
  __syncthreads();
  for (int s = 128; s > 0; s >>= 1) {
    if (tid < s) { sx[tid] += sx[tid + s]; sq[tid] += sq[tid + s]; sa[tid] += sa[tid + s]; }
    __syncthreads();
  }
  if (tid == 0) {
    float xem  = sx[0] / (float)M;
    float msem = sq[0] / (float)((size_t)Bn * KEEP * Dn);
    float accp = 100.f * (float)sa[0] / (float)M;
    out[0] = xem + msem;  // loss
    out[1] = xem;         // xe
    out[2] = msem;        // mse
    out[3] = accp;        // acc
  }
}

// ---------------- launcher ----------------
extern "C" void kernel_launch(void* const* d_in, const int* in_sizes, int n_in,
                              void* d_out, int out_size, void* d_ws, size_t ws_size,
                              hipStream_t stream) {
  const float* in_seq   = (const float*)d_in[0];
  const float* out_seq  = (const float*)d_in[1];
  const int*   drop_idx = (const int*)d_in[2];
  const int*   keep_idx = (const int*)d_in[3];
  float* out = (float*)d_out;

  char* ws = (char*)d_ws;
  unsigned short* Phi = (unsigned short*)(ws + OFF_PHI);
  unsigned short* Plo = (unsigned short*)(ws + OFF_PLO);
  unsigned short* Khi = (unsigned short*)(ws + OFF_KHI);
  unsigned short* Klo = (unsigned short*)(ws + OFF_KLO);
  float* rowmax       = (float*)(ws + OFF_RMAX);
  float* rowsum       = (float*)(ws + OFF_RSUM);
  float* tgt_logit    = (float*)(ws + OFF_TLOG);
  int*   tgt_col      = (int*)(ws + OFF_TCOL);
  float* mse_part     = (float*)(ws + OFF_MSEP);
  float* mpart        = (float*)(ws + OFF_MPRT);
  float* spart        = (float*)(ws + OFF_SPRT);

  convert_keys<<<(N * Kd / 4) / 256, 256, 0, stream>>>(in_seq, Khi, Klo);
  gather_preds<<<M, 256, 0, stream>>>(out_seq, drop_idx, Phi, Plo, tgt_col);
  mse_kernel<<<MSE_BLOCKS, 256, 0, stream>>>(in_seq, out_seq, keep_idx, mse_part);

  dim3 grid(N / BN, M / BM);   // (128, 64)
  gemm_softmax<<<grid, 256, 0, stream>>>(Phi, Plo, Khi, Klo,
                                         mpart, spart, tgt_logit, tgt_col);
  reduce_rows<<<M, NBLK, 0, stream>>>(mpart, spart, rowmax, rowsum);
  finalize_kernel<<<1, 256, 0, stream>>>(rowmax, rowsum, tgt_logit, mse_part, out);
}
